// MultiScaleFeatureMerger_90177133347027
// MI455X (gfx1250) — compile-verified, matrix operand order checked by reading, not run on hardware
//
#include <hip/hip_runtime.h>
#include <math.h>

// ---------------------------------------------------------------------------
// MultiScaleFeatureMerger for MI455X (gfx1250, wave32, WMMA)
//
// Math note: softmax over a single key == 1.0, so attn == V projection.
// Wq/bq/Wk/bk are dead. Per block:
//   cross = (LN(aux) @ Wv + bv) @ Wo + bo ; x = LN(x + cross)
//   x = LN(x + gelu(x @ W1 + b1) @ W2 + b2)
// GEMMs run on v_wmma_f32_16x16x32_bf16 (bf16 inputs, f32 accum), with
// double-buffered LDS tiles and async global->LDS DMA for the A operand.
// ---------------------------------------------------------------------------

#define BDIM 8192
#define DDIM 1024
#define ADIM 4

typedef __bf16 bf16_t;
typedef bf16_t v16bf __attribute__((ext_vector_type(16)));
typedef float  v8f   __attribute__((ext_vector_type(8)));
typedef int    v4i_g __attribute__((vector_size(16)));   // int4 for async-LDS builtin

union Frag { v16bf v; uint4 q[2]; };

// Native f32 -> bf16 (RNE); lets the backend use hardware cvt (pk) forms
// instead of an integer bit-twiddle chain.
__device__ __forceinline__ unsigned short f2bf(float f) {
    bf16_t h = (bf16_t)f;
    return __builtin_bit_cast(unsigned short, h);
}

// --- gfx1250 async global->LDS path (guarded; fallback = reg-staged copy) ---
#if defined(__has_builtin)
#  if __has_builtin(__builtin_amdgcn_global_load_async_to_lds_b128)
#    define MSFM_ASYNC_LDS 1
#  endif
#endif
#ifndef MSFM_ASYNC_LDS
#  define MSFM_ASYNC_LDS 0
#endif

#if MSFM_ASYNC_LDS
typedef __attribute__((address_space(1))) v4i_g* msfm_gptr;
typedef __attribute__((address_space(3))) v4i_g* msfm_lptr;
#  if __has_builtin(__builtin_amdgcn_s_wait_asynccnt)
#    define MSFM_WAIT_ASYNC() __builtin_amdgcn_s_wait_asynccnt(0)
#  else
#    define MSFM_WAIT_ASYNC() asm volatile("s_wait_asynccnt 0" ::: "memory")
#  endif
#else
#  define MSFM_WAIT_ASYNC() ((void)0)
#endif

// ---------------------------------------------------------------------------
// Fused (residual-add +) LayerNorm. One wave32 per row of D=1024.
// Writes f32 and/or packed-bf16 outputs.
// ---------------------------------------------------------------------------
__global__ __launch_bounds__(256)
void msfm_ln_fused(const float* __restrict__ x, const float* __restrict__ resid,
                   const float* __restrict__ g, const float* __restrict__ beta,
                   float* __restrict__ outF, unsigned short* __restrict__ outB)
{
    const int lane = threadIdx.x & 31;
    const int wave = threadIdx.x >> 5;
    const int row  = blockIdx.x * 8 + wave;
    const size_t base = (size_t)row * DDIM;

    float4 vals[8];
    float s = 0.f, s2 = 0.f;
#pragma unroll
    for (int j = 0; j < 8; ++j) {
        const int c4 = lane + j * 32;                 // float4 index in row
        float4 v = *reinterpret_cast<const float4*>(x + base + (size_t)c4 * 4);
        if (resid) {
            float4 r = *reinterpret_cast<const float4*>(resid + base + (size_t)c4 * 4);
            v.x += r.x; v.y += r.y; v.z += r.z; v.w += r.w;
        }
        vals[j] = v;
        s  += v.x + v.y + v.z + v.w;
        s2 += v.x * v.x + v.y * v.y + v.z * v.z + v.w * v.w;
    }
#pragma unroll
    for (int m = 16; m >= 1; m >>= 1) {
        s  += __shfl_xor(s,  m, 32);
        s2 += __shfl_xor(s2, m, 32);
    }
    const float mean = s * (1.0f / DDIM);
    const float var  = s2 * (1.0f / DDIM) - mean * mean;
    const float rstd = rsqrtf(var + 1e-5f);

#pragma unroll
    for (int j = 0; j < 8; ++j) {
        const int c4 = lane + j * 32;
        const float4 gg = *reinterpret_cast<const float4*>(g    + (size_t)c4 * 4);
        const float4 bb = *reinterpret_cast<const float4*>(beta + (size_t)c4 * 4);
        float4 v = vals[j];
        v.x = (v.x - mean) * rstd * gg.x + bb.x;
        v.y = (v.y - mean) * rstd * gg.y + bb.y;
        v.z = (v.z - mean) * rstd * gg.z + bb.z;
        v.w = (v.w - mean) * rstd * gg.w + bb.w;
        if (outF) *reinterpret_cast<float4*>(outF + base + (size_t)c4 * 4) = v;
        if (outB) {
            uint2 p;
            p.x = (unsigned)f2bf(v.x) | ((unsigned)f2bf(v.y) << 16);
            p.y = (unsigned)f2bf(v.z) | ((unsigned)f2bf(v.w) << 16);
            *reinterpret_cast<uint2*>(outB + base + (size_t)c4 * 4) = p;
        }
    }
}

// ---------------------------------------------------------------------------
// GEMM: out[B x N] = A_bf16[B x K] @ W_f32[K x N] + bias
// 128x128 tile, BK=64, 8 waves as 4(M) x 2(N); wave tile 32x64 = 2x4 WMMA.
// Double-buffered LDS; A tile via async global->LDS DMA, W tile staged in
// VGPRs (loads overlap the WMMA phase), converted f32->bf16, stored
// transposed [n][k] so A/B fragment reads are identical ds_read_b128 pairs.
// EPI: 0 = bf16 out, 1 = f32 out, 2 = exact-GELU -> bf16 out
// ---------------------------------------------------------------------------
template <int EPI>
__global__ __launch_bounds__(256)
void msfm_gemm_wmma(const unsigned short* __restrict__ A,
                    const float* __restrict__ W,
                    const float* __restrict__ bias,
                    float* __restrict__ outF,
                    unsigned short* __restrict__ outB,
                    int K, int N)
{
    __shared__ unsigned short As[2][128][64];   // 2 x 16 KB
    __shared__ unsigned short Bs[2][128][64];   // 2 x 16 KB (W transposed [n][k])

    const int tid  = threadIdx.x;
    const int lane = tid & 31;
    const int wave = tid >> 5;
    const int wm   = wave & 3;               // 4 row-waves
    const int wn   = wave >> 2;              // 2 col-waves
    const int lh   = lane >> 4;              // lane half (K-phase select)
    const int ll   = lane & 15;
    const int m0   = blockIdx.x * 128;
    const int n0   = blockIdx.y * 128;

    v8f acc[2][4];
#pragma unroll
    for (int i = 0; i < 2; ++i)
#pragma unroll
        for (int j = 0; j < 4; ++j)
#pragma unroll
            for (int r = 0; r < 8; ++r) acc[i][j][r] = 0.f;

    uint4  at[4];   // A staging (fallback path only)
    float4 wt[8];   // W staging

    // --- issue A tile loads for buffer `buf`, k-slab `k0` -------------------
    auto issueA = [&](int buf, int k0) {
#if MSFM_ASYNC_LDS
#pragma unroll
        for (int it = 0; it < 4; ++it) {
            const int idx = tid + it * 256;
            const int r = idx >> 3, c = idx & 7;
            __builtin_amdgcn_global_load_async_to_lds_b128(
                (msfm_gptr)(A + (size_t)(m0 + r) * K + k0 + c * 8),
                (msfm_lptr)&As[buf][r][c * 8],
                0, 0);
        }
#else
#pragma unroll
        for (int it = 0; it < 4; ++it) {
            const int idx = tid + it * 256;
            at[it] = *reinterpret_cast<const uint4*>(
                A + (size_t)(m0 + (idx >> 3)) * K + k0 + (idx & 7) * 8);
        }
#endif
        (void)buf;
    };
    auto commitA = [&](int buf) {
#if !MSFM_ASYNC_LDS
#pragma unroll
        for (int it = 0; it < 4; ++it) {
            const int idx = tid + it * 256;
            *reinterpret_cast<uint4*>(&As[buf][idx >> 3][(idx & 7) * 8]) = at[it];
        }
#endif
        (void)buf;
    };

    // --- issue / commit W tile (64 x 128 f32 -> bf16 transposed) ------------
    auto issueW = [&](int k0) {
#pragma unroll
        for (int it = 0; it < 8; ++it) {
            const int idx = tid + it * 256;
            wt[it] = *reinterpret_cast<const float4*>(
                W + (size_t)(k0 + (idx >> 5)) * N + n0 + (idx & 31) * 4);
        }
    };
    auto commitW = [&](int buf) {
#pragma unroll
        for (int it = 0; it < 8; ++it) {
            const int idx = tid + it * 256;
            const int kr = idx >> 5, c4 = idx & 31;
            Bs[buf][c4 * 4 + 0][kr] = f2bf(wt[it].x);
            Bs[buf][c4 * 4 + 1][kr] = f2bf(wt[it].y);
            Bs[buf][c4 * 4 + 2][kr] = f2bf(wt[it].z);
            Bs[buf][c4 * 4 + 3][kr] = f2bf(wt[it].w);
        }
    };

    // --- WMMA compute on buffer `buf` ---------------------------------------
    auto compute = [&](int buf) {
#pragma unroll
        for (int kk = 0; kk < 64; kk += 32) {
            Frag af[2], bfr[4];
#pragma unroll
            for (int i = 0; i < 2; ++i) {
                const unsigned short* p = &As[buf][wm * 32 + i * 16 + ll][kk + lh * 8];
                af[i].q[0] = *reinterpret_cast<const uint4*>(p);
                af[i].q[1] = *reinterpret_cast<const uint4*>(p + 16);
            }
#pragma unroll
            for (int j = 0; j < 4; ++j) {
                const unsigned short* p = &Bs[buf][wn * 64 + j * 16 + ll][kk + lh * 8];
                bfr[j].q[0] = *reinterpret_cast<const uint4*>(p);
                bfr[j].q[1] = *reinterpret_cast<const uint4*>(p + 16);
            }
#pragma unroll
            for (int i = 0; i < 2; ++i)
#pragma unroll
                for (int j = 0; j < 4; ++j)
                    acc[i][j] = __builtin_amdgcn_wmma_f32_16x16x32_bf16(
                        false, af[i].v, false, bfr[j].v,
                        (short)0, acc[i][j], false, false);
        }
    };

    // --- pipelined main loop ------------------------------------------------
    issueA(0, 0);
    issueW(0);
    commitA(0);
    commitW(0);
    MSFM_WAIT_ASYNC();
    __syncthreads();

    int cur = 0;
    for (int k0 = 0; k0 < K; k0 += 64) {
        const bool more = (k0 + 64) < K;
        if (more) {
            issueA(cur ^ 1, k0 + 64);   // DMA / loads in flight during compute
            issueW(k0 + 64);
        }
        compute(cur);
        if (more) {
            commitA(cur ^ 1);
            commitW(cur ^ 1);
        }
        MSFM_WAIT_ASYNC();
        __syncthreads();
        cur ^= 1;
    }

    // --- epilogue: VGPR r -> M = r (lanes 0-15) / 8+r (lanes 16-31) --------
#pragma unroll
    for (int i = 0; i < 2; ++i) {
#pragma unroll
        for (int j = 0; j < 4; ++j) {
            const int col = n0 + wn * 64 + j * 16 + ll;
            const float bz = bias[col];
#pragma unroll
            for (int r = 0; r < 8; ++r) {
                const int row = m0 + wm * 32 + i * 16 + lh * 8 + r;
                float v = acc[i][j][r] + bz;
                if (EPI == 2) v = 0.5f * v * (1.0f + erff(v * 0.70710678118654752f));
                if (EPI == 1) outF[(size_t)row * N + col] = v;
                else          outB[(size_t)row * N + col] = f2bf(v);
            }
        }
    }
}

// ---------------------------------------------------------------------------
extern "C" void kernel_launch(void* const* d_in, const int* in_sizes, int n_in,
                              void* d_out, int out_size, void* d_ws, size_t ws_size,
                              hipStream_t stream)
{
    (void)in_sizes; (void)n_in; (void)out_size; (void)ws_size;

    const float* main_f    = (const float*)d_in[0];
    const float* aux       = (const float*)d_in[1];
    const float* norm_g    = (const float*)d_in[2];
    const float* norm_b    = (const float*)d_in[3];
    const float* ln_aux_g  = (const float*)d_in[4];
    const float* ln_aux_b  = (const float*)d_in[5];
    const float* ln_main_g = (const float*)d_in[6];
    const float* ln_main_b = (const float*)d_in[7];
    const float* ln_ffn_g  = (const float*)d_in[8];
    const float* ln_ffn_b  = (const float*)d_in[9];
    // d_in[10..13] = Wq, bq, Wk, bk : dead (softmax over 1 key == 1.0)
    const float* Wv = (const float*)d_in[14];
    const float* bv = (const float*)d_in[15];
    const float* Wo = (const float*)d_in[16];
    const float* bo = (const float*)d_in[17];
    const float* W1 = (const float*)d_in[18];
    const float* b1 = (const float*)d_in[19];
    const float* W2 = (const float*)d_in[20];
    const float* b2 = (const float*)d_in[21];
    float* out = (float*)d_out;

    char* ws = (char*)d_ws;
    size_t off = 0;
    float*          xF   = (float*)(ws + off);          off += (size_t)BDIM * DDIM * 4;
    unsigned short* xB   = (unsigned short*)(ws + off); off += (size_t)BDIM * DDIM * 2;
    unsigned short* auxB = (unsigned short*)(ws + off); off += (size_t)BDIM * DDIM * 2;
    unsigned short* vB   = (unsigned short*)(ws + off); off += (size_t)BDIM * DDIM * 2;
    float*          tmpF = (float*)(ws + off);          off += (size_t)BDIM * DDIM * 4;
    unsigned short* hB   = (unsigned short*)(ws + off); // B x 4D bf16

    const dim3 blk(256);
    const dim3 lnGrid(BDIM / 8);
    const dim3 gD(BDIM / 128, DDIM / 128);       // N = 1024
    const dim3 gH(BDIM / 128, 4 * DDIM / 128);   // N = 4096

    // x = LN(main)
    msfm_ln_fused<<<lnGrid, blk, 0, stream>>>(main_f, nullptr, norm_g, norm_b, xF, xB);

    for (int a = 0; a < ADIM; ++a) {
        const size_t dd = (size_t)DDIM * DDIM;
        // aux_ln (bf16 only — feeds GEMM)
        msfm_ln_fused<<<lnGrid, blk, 0, stream>>>(
            aux + (size_t)a * BDIM * DDIM, nullptr,
            ln_aux_g + a * DDIM, ln_aux_b + a * DDIM, nullptr, auxB);
        // v = aux_ln @ Wv + bv  (bf16 out)
        msfm_gemm_wmma<0><<<gD, blk, 0, stream>>>(
            auxB, Wv + a * dd, bv + a * DDIM, nullptr, vB, DDIM, DDIM);
        // cross = v @ Wo + bo  (f32 out)
        msfm_gemm_wmma<1><<<gD, blk, 0, stream>>>(
            vB, Wo + a * dd, bo + a * DDIM, tmpF, nullptr, DDIM, DDIM);
        // x = LN(x + cross)
        msfm_ln_fused<<<lnGrid, blk, 0, stream>>>(
            xF, tmpF, ln_main_g + a * DDIM, ln_main_b + a * DDIM, xF, xB);
        // h = gelu(x @ W1 + b1)  (bf16 out)
        msfm_gemm_wmma<2><<<gH, blk, 0, stream>>>(
            xB, W1 + (size_t)a * DDIM * 4 * DDIM, b1 + (size_t)a * 4 * DDIM,
            nullptr, hB, DDIM, 4 * DDIM);
        // ffn = h @ W2 + b2  (f32 out)
        msfm_gemm_wmma<1><<<gD, blk, 0, stream>>>(
            hB, W2 + (size_t)a * 4 * DDIM * DDIM, b2 + a * DDIM,
            tmpF, nullptr, 4 * DDIM, DDIM);
        // x = LN(x + ffn)  (final block writes d_out)
        float* dstF = (a == ADIM - 1) ? out : xF;
        msfm_ln_fused<<<lnGrid, blk, 0, stream>>>(
            xF, tmpF, ln_ffn_g + a * DDIM, ln_ffn_b + a * DDIM, dstF, xB);
    }
}
